// WindowAttention_26749056319837
// MI455X (gfx1250) — matrix operand order repaired and checked by reading, not compiled, not run on hardware
//
#include <hip/hip_runtime.h>
#include <hip/hip_bf16.h>

// ---------------------------------------------------------------------------
// Fused Swin window-attention for gfx1250 (MI455X), wave32 + WMMA f16.
//   qkv = x @ qkv_w^T + qkv_b          (WMMA, f16 operands, f32 acc)
//   S   = (q*scale) @ k^T + rpb + mask (WMMA + f32 epilogue)
//   P   = softmax(S)                   (f32, register-resident rows)
//   o   = P @ v                        (WMMA)
//   out = o @ proj_w^T + proj_b        (WMMA, f32 stores)
// One block per window (4096 blocks, 256 threads = 8 wave32).
// ---------------------------------------------------------------------------

typedef __attribute__((ext_vector_type(16))) _Float16 v16h;
typedef __attribute__((ext_vector_type(8)))  _Float16 v8h;
typedef __attribute__((ext_vector_type(2)))  _Float16 h2;
typedef __attribute__((ext_vector_type(8)))  float    v8f;

#define NTOK   49      // window tokens (7x7)
#define NPAD   64      // padded to 4 x 16 tiles
#define CDIM   128
#define HEADS  4
#define HDIM   32
#define QK_SCALE 0.17677669529663689f   // 32^-0.5

__device__ __forceinline__ v8f wmma_f16(v16h a, v16h b, v8f c) {
    // D = A(16x32 f16) * B(32x16 f16) + C(16x16 f32)
    return __builtin_amdgcn_wmma_f32_16x16x32_f16(
        false, a, false, b, (short)0, c, false, false);
}

// A fragment: 16x32 tile from row-major LDS [rows][ld] at (m0, k0).
// Lane layout: M = lane%16; VGPR v holds K pairs
//   K = (lane/16)*16 + v*2 + {0,1}
__device__ __forceinline__ v16h load_A(const _Float16* __restrict__ s,
                                       int ld, int m0, int k0, int lane) {
    const _Float16* row = s + (m0 + (lane & 15)) * ld;
    int kb = k0 + ((lane >> 4) & 1) * 16;
    union { v16h v; h2 p[8]; } ua;
#pragma unroll
    for (int vv = 0; vv < 8; ++vv) {
        int k = kb + (vv << 1);
        ua.p[vv] = *(const h2*)(row + k);          // 32-bit ds loads
    }
    return ua.v;
}

// B fragment: 32x16 tile. LDS holds B transposed: sT[n][k], row stride ld.
// Lane layout: N = lane%16, K = (lane/16)*16 + e  -> 32 contiguous bytes/lane.
__device__ __forceinline__ v16h load_B(const _Float16* __restrict__ sT,
                                       int ld, int k0, int n0, int lane) {
    const _Float16* p = sT + (n0 + (lane & 15)) * ld
                           + k0 + ((lane >> 4) & 1) * 16;
    return *(const v16h*)p;                        // ds_load_b128 x2
}

// One 128-feature slice of the QKV projection; P in {0:Q, 1:K, 2:V} is a
// compile-time constant so the epilogue store target has no branches.
template <int P>
__device__ __forceinline__ void qkv_slice(const float* __restrict__ qkv_w,
                                          const float* __restrict__ qkv_b,
                                          const _Float16* __restrict__ xs,
                                          _Float16* __restrict__ wbuf,
                                          _Float16* __restrict__ Qs,
                                          _Float16* __restrict__ Ks,
                                          _Float16* __restrict__ Vs,
                                          int tid, int lane, int wv) {
    const float* wg = qkv_w + P * CDIM * CDIM;     // rows are already B^T
#pragma unroll
    for (int j = 0; j < 16; ++j) {                 // stage 128x128 f32->f16
        int f = tid + j * 256;                     // float4 idx, 0..4095
        float4 v = *(const float4*)(wg + f * 4);
        _Float16* d = wbuf + f * 4;
        d[0] = (_Float16)v.x; d[1] = (_Float16)v.y;
        d[2] = (_Float16)v.z; d[3] = (_Float16)v.w;
    }
    __syncthreads();

#pragma unroll
    for (int it = 0; it < 4; ++it) {               // 32 tiles over 8 waves
        int t  = wv + it * 8;
        int m0 = (t & 3) << 4, n0 = (t >> 2) << 4; // m0 is wave-invariant
        v8f acc = {};
#pragma unroll
        for (int kc = 0; kc < 4; ++kc) {
            v16h a  = load_A(xs,   CDIM, m0, kc * 32, lane);
            v16h bm = load_B(wbuf, CDIM, kc * 32, n0, lane);
            acc = wmma_f16(a, bm, acc);
        }
        int col  = n0 + (lane & 15);               // local feature 0..127
        float bv = qkv_b[P * CDIM + col];
        int h = col >> 5, dd = col & 31;
        int mbase = m0 + ((lane >> 4) << 3);
        if (P == 2) {
            // V^T rows: 8 consecutive m -> one 16B-aligned ds_store_b128
            v8h vv;
#pragma unroll
            for (int r = 0; r < 8; ++r) vv[r] = (_Float16)(acc[r] + bv);
            *(v8h*)(Vs + (h * HDIM + dd) * NPAD + mbase) = vv;
        } else {
#pragma unroll
            for (int r = 0; r < 8; ++r) {
                float val = acc[r] + bv;
                if (P == 0)
                    Qs[(h * NPAD + mbase + r) * HDIM + dd] =
                        (_Float16)(val * QK_SCALE);
                else
                    Ks[(h * NPAD + mbase + r) * HDIM + dd] = (_Float16)val;
            }
        }
    }
    __syncthreads();
}

__global__ void __launch_bounds__(256)
win_attn_fused(const float* __restrict__ x,      // (4096, 49, 128)
               const float* __restrict__ mask,   // (64, 49, 49)
               const float* __restrict__ qkv_w,  // (384, 128)
               const float* __restrict__ qkv_b,  // (384,)
               const float* __restrict__ rpb,    // (169, 4)
               const float* __restrict__ proj_w, // (128, 128)
               const float* __restrict__ proj_b, // (128,)
               float* __restrict__ out)          // (4096, 49, 128)
{
    const int b    = blockIdx.x;
    const int tid  = threadIdx.x;
    const int lane = tid & 31;
    const int wv   = tid >> 5;     // wave id, 0..7

    // ---- dynamic LDS, 131072 B, time-multiplexed regions -----------------
    extern __shared__ unsigned char smem[];
    _Float16* xs   = (_Float16*)smem;                    // [64][128] f16 (x), later y
    _Float16* ys   = xs;                                 // o staged for proj
    _Float16* wbuf = (_Float16*)(smem + 16384);          // [128][128] f16 qkv_w slice
    float*    Ssc  = (float*)   (smem + 16384);          // [4][64][64] f32 scores
    _Float16* wp   = (_Float16*)(smem + 16384);          // [128][128] f16 proj_w
    _Float16* Qs   = (_Float16*)(smem + 16384 + 65536);  // [4][64][32]
    _Float16* Ks   = Qs + HEADS * NPAD * HDIM;           // [4][64][32]
    _Float16* Ps   = Qs;                                 // [4][64][64] aliases Q+K
    _Float16* Vs   = (_Float16*)(smem + 16384 + 65536 + 32768); // [4][32][64] = V^T

    // ---- Phase 1: stage x tile (f32 -> f16), zero-pad rows 49..63 --------
    {
        const float* xg = x + (size_t)b * (NTOK * CDIM);
#pragma unroll
        for (int j = 0; j < 8; ++j) {
            int f = tid + j * 256;          // float4 index, 0..2047
            int m = f >> 5;
            int c = (f & 31) << 2;
            _Float16* d = xs + m * CDIM + c;
            if (m < NTOK) {
                float4 v = *(const float4*)(xg + m * CDIM + c);
                d[0] = (_Float16)v.x; d[1] = (_Float16)v.y;
                d[2] = (_Float16)v.z; d[3] = (_Float16)v.w;
            } else {
                d[0] = d[1] = d[2] = d[3] = (_Float16)0.f;
            }
        }
    }
    __syncthreads();

    // ---- Phase 2: QKV GEMM, three statically-specialized slices ----------
    qkv_slice<0>(qkv_w, qkv_b, xs, wbuf, Qs, Ks, Vs, tid, lane, wv);
    qkv_slice<1>(qkv_w, qkv_b, xs, wbuf, Qs, Ks, Vs, tid, lane, wv);
    qkv_slice<2>(qkv_w, qkv_b, xs, wbuf, Qs, Ks, Vs, tid, lane, wv);

    // ---- Phase 3: scores S = (q*scale) @ k^T, K-depth 32 = one WMMA ------
#pragma unroll
    for (int it = 0; it < 8; ++it) {               // 4 heads x 4x4 tiles
        int t = wv + it * 8;
        int h = t >> 4, m0 = ((t >> 2) & 3) << 4, n0 = (t & 3) << 4;
        v16h a  = load_A(Qs + h * NPAD * HDIM, HDIM, m0, 0, lane);
        v16h bm = load_B(Ks + h * NPAD * HDIM, HDIM, 0, n0, lane);
        v8f acc = {};
        acc = wmma_f16(a, bm, acc);
        int n = n0 + (lane & 15);
        int mbase = m0 + ((lane >> 4) << 3);
#pragma unroll
        for (int r = 0; r < 8; ++r)
            Ssc[(h * NPAD + mbase + r) * NPAD + n] = acc[r];
    }
    __syncthreads();

    // ---- Phase 4: +rpb +mask, f32 softmax (1 thread per head-row) --------
    if (tid < HEADS * NTOK) {
        int h = tid / NTOK, n = tid - h * NTOK;
        int ni = n / 7, nj = n - ni * 7;
        const float* mrow = mask + (((b & 63) * NTOK) + n) * NTOK;
        const float* srow = Ssc + (h * NPAD + n) * NPAD;
        float buf[NTOK];
        float mx = -3.0e38f;
#pragma unroll
        for (int m2 = 0; m2 < NTOK; ++m2) {
            int rel = (ni - m2 / 7 + 6) * 13 + (nj - m2 % 7 + 6);
            float s = srow[m2] + rpb[rel * HEADS + h] + mrow[m2];
            buf[m2] = s;
            mx = fmaxf(mx, s);
        }
        float sum = 0.f;
#pragma unroll
        for (int m2 = 0; m2 < NTOK; ++m2) {
            float e = __expf(buf[m2] - mx);
            buf[m2] = e;
            sum += e;
        }
        float inv = 1.0f / sum;
        _Float16* prow = Ps + (h * NPAD + n) * NPAD;
#pragma unroll
        for (int m2 = 0; m2 < NTOK; ++m2) prow[m2] = (_Float16)(buf[m2] * inv);
#pragma unroll
        for (int m2 = NTOK; m2 < NPAD; ++m2) prow[m2] = (_Float16)0.f;
    } else {
        int t2 = tid - HEADS * NTOK;                 // 0..59 -> 60 pad rows
        int h = t2 / 15, m = NTOK + (t2 - (t2 / 15) * 15);
        _Float16* prow = Ps + (h * NPAD + m) * NPAD;
#pragma unroll
        for (int m2 = 0; m2 < NPAD; ++m2) prow[m2] = (_Float16)0.f;
    }
    __syncthreads();

    // ---- Phase 5: o = P @ v (K-depth 64), plus proj_w staging ------------
    {
#pragma unroll
        for (int j = 0; j < 16; ++j) {               // stage proj_w f32->f16
            int f = tid + j * 256;
            float4 v = *(const float4*)(proj_w + f * 4);
            _Float16* d = wp + f * 4;
            d[0] = (_Float16)v.x; d[1] = (_Float16)v.y;
            d[2] = (_Float16)v.z; d[3] = (_Float16)v.w;
        }
#pragma unroll
        for (int it = 0; it < 4; ++it) {             // 4 heads x 4x2 tiles
            int t = wv + it * 8;
            int h = t >> 3, rem = t & 7;
            int m0 = (rem >> 1) << 4, n0 = (rem & 1) << 4;
            v8f acc = {};
#pragma unroll
            for (int kc = 0; kc < 2; ++kc) {
                v16h a  = load_A(Ps + h * NPAD * NPAD, NPAD, m0, kc * 32, lane);
                v16h bm = load_B(Vs + h * HDIM * NPAD, NPAD, kc * 32, n0, lane);
                acc = wmma_f16(a, bm, acc);
            }
            int cc = h * HDIM + n0 + (lane & 15);    // output channel
            int mbase = m0 + ((lane >> 4) << 3);
#pragma unroll
            for (int r = 0; r < 8; ++r)
                ys[(mbase + r) * CDIM + cc] = (_Float16)acc[r];
        }
    }
    __syncthreads();

    // ---- Phase 6: out = y @ proj_w^T + proj_b, f32 stores ----------------
    {
        float* og = out + (size_t)b * (NTOK * CDIM);
#pragma unroll
        for (int it = 0; it < 4; ++it) {
            int t = wv + it * 8;
            int m0 = (t & 3) << 4, n0 = (t >> 2) << 4;
            v8f acc = {};
#pragma unroll
            for (int kc = 0; kc < 4; ++kc) {
                v16h a  = load_A(ys, CDIM, m0, kc * 32, lane);
                v16h bm = load_B(wp, CDIM, kc * 32, n0, lane);
                acc = wmma_f16(a, bm, acc);
            }
            int n = n0 + (lane & 15);
            float bv = proj_b[n];
            int mbase = m0 + ((lane >> 4) << 3);
#pragma unroll
            for (int r = 0; r < 8; ++r) {
                int m = mbase + r;
                if (m < NTOK) og[m * CDIM + n] = acc[r] + bv;
            }
        }
    }
}

extern "C" void kernel_launch(void* const* d_in, const int* in_sizes, int n_in,
                              void* d_out, int out_size, void* d_ws, size_t ws_size,
                              hipStream_t stream) {
    const float* x      = (const float*)d_in[0];
    const float* mask   = (const float*)d_in[1];
    const float* qkv_w  = (const float*)d_in[2];
    const float* qkv_b  = (const float*)d_in[3];
    const float* rpb    = (const float*)d_in[4];
    const float* proj_w = (const float*)d_in[5];
    const float* proj_b = (const float*)d_in[6];
    float* out = (float*)d_out;

    // 4096 windows, one workgroup (8 wave32) each, 128 KB dynamic LDS.
    win_attn_fused<<<4096, 256, 131072, stream>>>(
        x, mask, qkv_w, qkv_b, rpb, proj_w, proj_b, out);
}
